// AttentionMILModel_2087354106714
// MI455X (gfx1250) — matrix operand order repaired and checked by reading, not compiled, not run on hardware
//
#include <hip/hip_runtime.h>
#include <hip/hip_bf16.h>

typedef __attribute__((ext_vector_type(16))) __bf16 v16bf;
typedef __attribute__((ext_vector_type(8)))  float  v8f;
typedef __attribute__((ext_vector_type(4)))  unsigned int u32x4;
typedef __attribute__((ext_vector_type(8)))  int  i32x8;
typedef __attribute__((ext_vector_type(4)))  int  i32x4;

#define N_PATCHES 100000
#define N_BAGS    32
#define D_IN      1024
#define D_FEAT    512
#define D_ATTN    256
#define N_CLASSES 2
#define BS_CHUNKS 13

__device__ __forceinline__ unsigned short f2bf(float f) {
  unsigned int u = __float_as_uint(f);
  unsigned int r = u + 0x7FFFu + ((u >> 16) & 1u);   // round-to-nearest-even
  return (unsigned short)(r >> 16);
}

// transpose + convert: src[K,N] f32 -> dst[N,K] bf16
__global__ void cvt_transpose_bf16_kernel(const float* __restrict__ src,
                                          unsigned short* __restrict__ dst,
                                          int K, int N) {
  int idx = blockIdx.x * blockDim.x + threadIdx.x;
  if (idx < K * N) {
    int k = idx / N, n = idx % N;
    dst[(size_t)n * K + k] = f2bf(src[idx]);
  }
}

// TDM: load a 128(row) x 32(col) bf16 tile from a row-major [Nmat,K] bf16 matrix
// into LDS at byte offset lds_off, padding each 64B row with 16B (-> 80B stride).
__device__ __forceinline__ void tdm_load_b_tile(const unsigned short* tile_ptr,
                                                int K, int Nmat, unsigned lds_off) {
  unsigned long long ga = (unsigned long long)(size_t)tile_ptr;
  u32x4 g0;
  g0[0] = 1u;                                            // count=1, user descriptor
  g0[1] = lds_off;                                       // lds_addr
  g0[2] = (unsigned)(ga & 0xFFFFFFFFull);                // global_addr lo
  g0[3] = (unsigned)((ga >> 32) & 0x01FFFFFFull) | 0x80000000u;  // addr hi | type=2
  i32x8 g1;
  // data_size=2B(1<<16), pad_enable(1<<20), pad_interval=16dw(3<<22), pad_amount=4dw(3<<25)
  g1[0] = (int)((1u << 16) | (1u << 20) | (3u << 22) | (3u << 25));
  g1[1] = (int)(((unsigned)K & 0xFFFFu) << 16);          // tensor_dim0[15:0] @ bits63:48
  g1[2] = (int)((((unsigned)K >> 16) & 0xFFFFu) |        // tensor_dim0[31:16]
                (((unsigned)Nmat & 0xFFFFu) << 16));     // tensor_dim1[15:0]
  g1[3] = (int)((((unsigned)Nmat >> 16) & 0xFFFFu) |     // tensor_dim1[31:16]
                (32u << 16));                            // tile_dim0 = 32
  g1[4] = (int)128u;                                     // tile_dim1 = 128, tile_dim2 = 0
  g1[5] = (int)K;                                        // tensor_dim0_stride lo32
  g1[6] = 0;                                             // stride hi, dim1_stride lo (unused 2D)
  g1[7] = 0;
  i32x4 z4 = {0, 0, 0, 0};
#if __clang_major__ >= 23
  i32x8 z8 = {0, 0, 0, 0, 0, 0, 0, 0};
  __builtin_amdgcn_tensor_load_to_lds(g0, g1, z4, z4, z8, 0);
#else
  __builtin_amdgcn_tensor_load_to_lds(g0, g1, z4, z4, 0);
#endif
}

union FragCast { uint4 u[2]; v16bf v; };

// Out[M,N] = act(A[M,K](f32) @ B[K,N] + bias[N]), B given pre-transposed bf16 [N,K].
// EPI: 0=ReLU, 1=tanh. Compile-time M/K/N -> immediate-offset addressing.
// Block tile 128x128, BK=32, double-buffered LDS, B tiles via Tensor Data Mover,
// 8 waves x (2x4) WMMA tiles.
template<int EPI, int M, int K, int N>
__global__ __launch_bounds__(256) void gemm_bias_act_kernel(
    const float* __restrict__ A, const unsigned short* __restrict__ Bt,
    const float* __restrict__ bias, float* __restrict__ Out) {
  // 80-byte row stride: 16B-aligned b128 reads, conflict-free 20-bank stride.
  __shared__ unsigned short As[2][128][40];
  __shared__ unsigned short Bs[2][128][40];

  const int tid  = threadIdx.x;
  const int lane = tid & 31;
  const int wave = tid >> 5;
  const int wm   = wave & 3;
  const int wn   = wave >> 2;
  const int l15  = lane & 15;
  const int lh   = lane >> 4;
  const int mBase = blockIdx.x * 128;
  const int nBase = blockIdx.y * 128;

  v8f acc[2][4];
#pragma unroll
  for (int mt = 0; mt < 2; ++mt)
#pragma unroll
    for (int nt = 0; nt < 4; ++nt)
      acc[mt][nt] = (v8f){0.f, 0.f, 0.f, 0.f, 0.f, 0.f, 0.f, 0.f};

  constexpr int NK = K >> 5;
  const int row0 = tid >> 3;           // 0..31 (step 32 per i)
  const int c40  = (tid & 7) << 2;

  auto loadA = [&](int k0, int buf) {
#pragma unroll
    for (int i = 0; i < 4; ++i) {
      int row = row0 + i * 32;
      int gr  = mBase + row;
      gr = gr < M ? gr : M - 1;                 // clamp: no divergent branch
      const float* ap = A + (size_t)gr * K + k0 + c40;
      float4 v = *(const float4*)ap;
      __builtin_prefetch(ap + 32, 0, 3);        // near-scope global_prefetch_b8
      unsigned lo = (unsigned)f2bf(v.x) | ((unsigned)f2bf(v.y) << 16);
      unsigned hi = (unsigned)f2bf(v.z) | ((unsigned)f2bf(v.w) << 16);
      *(uint2*)&As[buf][row][c40] = make_uint2(lo, hi);  // ds_store_b64
    }
  };
  auto issueB = [&](int k0, int buf) {
    if (wave == 0) {
      unsigned lds = (unsigned)(size_t)&Bs[buf][0][0];
      tdm_load_b_tile(Bt + (size_t)nBase * K + k0, K, N, lds);
    }
  };

  // prologue: stage k-step 0 into buffer 0
  loadA(0, 0);
  issueB(0, 0);
  if (wave == 0) __builtin_amdgcn_s_wait_tensorcnt(0);
  __syncthreads();

#pragma unroll 2
  for (int i = 0; i < NK; ++i) {
    const int cur = i & 1;
    if (i + 1 < NK) {                 // prefetch next k-step into other buffer
      loadA((i + 1) << 5, cur ^ 1);
      issueB((i + 1) << 5, cur ^ 1);
    }

    const unsigned short (*as)[40] = As[cur];
    const unsigned short (*bs)[40] = Bs[cur];

    v16bf afr[2];
#pragma unroll
    for (int mt = 0; mt < 2; ++mt) {
      int row = wm * 32 + mt * 16 + l15;
      int kb  = lh * 8;               // lanes 0-15: K 0..7,16..23 ; 16-31: +8
      FragCast fc;
      fc.u[0] = *(const uint4*)&as[row][kb];
      fc.u[1] = *(const uint4*)&as[row][kb + 16];
      afr[mt] = fc.v;
    }
    v16bf bfr[4];
#pragma unroll
    for (int nt = 0; nt < 4; ++nt) {
      int col = wn * 64 + nt * 16 + l15;
      FragCast fc;                    // lanes 0-15: K 0..7,16..23 ; 16-31: +8
      fc.u[0] = *(const uint4*)&bs[col][lh * 8];
      fc.u[1] = *(const uint4*)&bs[col][lh * 8 + 16];
      bfr[nt] = fc.v;
    }
#pragma unroll
    for (int mt = 0; mt < 2; ++mt)
#pragma unroll
      for (int nt = 0; nt < 4; ++nt)
        acc[mt][nt] = __builtin_amdgcn_wmma_f32_16x16x32_bf16(
            false, afr[mt], false, bfr[nt], (short)0, acc[mt][nt], false, false);

    if (wave == 0) __builtin_amdgcn_s_wait_tensorcnt(0);  // next B tile resident
    __syncthreads();
  }

  // epilogue: bias + activation + store (immediate-offset addressing)
  const int nCol = nBase + wn * 64 + l15;
  float bn[4];
#pragma unroll
  for (int nt = 0; nt < 4; ++nt) bn[nt] = bias[nCol + nt * 16];

  const int mRow0 = mBase + wm * 32 + lh * 8;
  float* obase = Out + (size_t)mRow0 * N + nCol;

  if (mBase + 128 <= M) {            // uniform fast path: full tile in range
#pragma unroll
    for (int mt = 0; mt < 2; ++mt)
#pragma unroll
      for (int nt = 0; nt < 4; ++nt)
#pragma unroll
        for (int r = 0; r < 8; ++r) {
          float v = acc[mt][nt][r] + bn[nt];
          v = (EPI == 0) ? fmaxf(v, 0.f) : tanhf(v);
          obase[(size_t)(mt * 16 + r) * N + nt * 16] = v;   // const offsets
        }
  } else {                           // ragged edge block only
#pragma unroll
    for (int mt = 0; mt < 2; ++mt)
#pragma unroll
      for (int nt = 0; nt < 4; ++nt)
#pragma unroll
        for (int r = 0; r < 8; ++r) {
          if (mRow0 + mt * 16 + r < M) {
            float v = acc[mt][nt][r] + bn[nt];
            v = (EPI == 0) ? fmaxf(v, 0.f) : tanhf(v);
            obase[(size_t)(mt * 16 + r) * N + nt * 16] = v;
          }
        }
  }
}

// scores[i] = dot(Sattn[i,:256], Wa2) + ba2 ; one wave per row, float4 loads
__global__ void scores_kernel(const float* __restrict__ Sattn,
                              const float* __restrict__ Wa2,
                              const float* __restrict__ ba2,
                              float* __restrict__ scores, int nrows) {
  int row  = blockIdx.x * 8 + (threadIdx.x >> 5);
  int lane = threadIdx.x & 31;
  if (row >= nrows) return;
  const float4* p4 = (const float4*)(Sattn + (size_t)row * D_ATTN);
  const float4* w4 = (const float4*)Wa2;
  float v = 0.f;
#pragma unroll
  for (int j = 0; j < 2; ++j) {
    float4 a = p4[lane * 2 + j];
    float4 b = w4[lane * 2 + j];
    v += a.x * b.x + a.y * b.y + a.z * b.z + a.w * b.w;
  }
#pragma unroll
  for (int o = 16; o > 0; o >>= 1) v += __shfl_xor(v, o, 32);
  if (lane == 0) scores[row] = v + ba2[0];
}

__global__ void offsets_kernel(const long long* __restrict__ bag_sizes,
                               int* __restrict__ offs, int nb) {
  if (threadIdx.x == 0) {
    int s = 0; offs[0] = 0;
    for (int i = 0; i < nb; ++i) { s += (int)bag_sizes[i]; offs[i + 1] = s; }
  }
}

__global__ void softmax_kernel(const float* __restrict__ scores,
                               const int* __restrict__ offs,
                               float* __restrict__ w) {
  __shared__ float red[256];
  __shared__ float s_max, s_sum;
  int b = blockIdx.x;
  int s0 = offs[b], s1 = offs[b + 1];
  int tid = threadIdx.x;
  float m = -3.4e38f;
  for (int i = s0 + tid; i < s1; i += 256) m = fmaxf(m, scores[i]);
  red[tid] = m; __syncthreads();
  for (int o = 128; o > 0; o >>= 1) {
    if (tid < o) red[tid] = fmaxf(red[tid], red[tid + o]);
    __syncthreads();
  }
  if (tid == 0) s_max = red[0];
  __syncthreads();
  float s = 0.f;
  for (int i = s0 + tid; i < s1; i += 256) s += expf(scores[i] - s_max);
  red[tid] = s; __syncthreads();
  for (int o = 128; o > 0; o >>= 1) {
    if (tid < o) red[tid] += red[tid + o];
    __syncthreads();
  }
  if (tid == 0) s_sum = red[0];
  __syncthreads();
  float inv = 1.f / s_sum;
  for (int i = s0 + tid; i < s1; i += 256) w[i] = expf(scores[i] - s_max) * inv;
}

// deterministic segment-weighted sum: fixed-order partials, then fixed-order reduce
__global__ void bagsum_partial_kernel(const float* __restrict__ H2,
                                      const float* __restrict__ w,
                                      const int* __restrict__ offs,
                                      float* __restrict__ partial) {
  int b = blockIdx.x, ch = blockIdx.y, col = threadIdx.x;
  int s0 = offs[b], s1 = offs[b + 1];
  int len = s1 - s0;
  int per = (len + (int)gridDim.y - 1) / (int)gridDim.y;
  int r0 = s0 + ch * per;
  int r1 = min(s1, r0 + per);
  float acc = 0.f;
  for (int r = r0; r < r1; ++r) acc += w[r] * H2[(size_t)r * D_FEAT + col];
  partial[((size_t)b * gridDim.y + ch) * D_FEAT + col] = acc;
}

__global__ void bagsum_reduce_kernel(const float* __restrict__ partial,
                                     float* __restrict__ bag_emb, int nch) {
  int b = blockIdx.x, col = threadIdx.x;
  float acc = 0.f;
  for (int c = 0; c < nch; ++c)
    acc += partial[((size_t)b * nch + c) * D_FEAT + col];
  bag_emb[b * D_FEAT + col] = acc;
}

__global__ void classifier_kernel(const float* __restrict__ bag_emb,
                                  const float* __restrict__ Wc,
                                  const float* __restrict__ bc,
                                  float* __restrict__ out) {
  int t = threadIdx.x;
  if (t >= N_BAGS * N_CLASSES) return;
  int b = t >> 1, c = t & 1;
  float acc = bc[c];
  for (int k = 0; k < D_FEAT; ++k) acc += bag_emb[b * D_FEAT + k] * Wc[k * N_CLASSES + c];
  out[b * N_CLASSES + c] = acc;
}

extern "C" void kernel_launch(void* const* d_in, const int* in_sizes, int n_in,
                              void* d_out, int out_size, void* d_ws, size_t ws_size,
                              hipStream_t stream) {
  (void)in_sizes; (void)n_in; (void)out_size; (void)ws_size;
  const float*     features  = (const float*)d_in[0];
  const long long* bag_sizes = (const long long*)d_in[1];
  const float* W1  = (const float*)d_in[2];
  const float* b1  = (const float*)d_in[3];
  const float* W2  = (const float*)d_in[4];
  const float* b2  = (const float*)d_in[5];
  const float* Wa1 = (const float*)d_in[6];
  const float* ba1 = (const float*)d_in[7];
  const float* Wa2 = (const float*)d_in[8];
  const float* ba2 = (const float*)d_in[9];
  const float* Wc  = (const float*)d_in[10];
  const float* bc  = (const float*)d_in[11];
  float* out = (float*)d_out;

  char* ws = (char*)d_ws;
  size_t off = 0;
  auto alloc = [&](size_t bytes) -> void* {
    off = (off + 255) & ~(size_t)255;
    void* p = ws + off;
    off += bytes;
    return p;
  };
  unsigned short* W1t  = (unsigned short*)alloc((size_t)D_IN   * D_FEAT * 2);  // [N,K] bf16
  unsigned short* W2t  = (unsigned short*)alloc((size_t)D_FEAT * D_FEAT * 2);
  unsigned short* Wa1t = (unsigned short*)alloc((size_t)D_FEAT * D_ATTN * 2);
  float* scores  = (float*)alloc((size_t)N_PATCHES * 4);
  float* attw    = (float*)alloc((size_t)N_PATCHES * 4);
  int*   offs    = (int*)alloc((N_BAGS + 1) * 4);
  float* bag_emb = (float*)alloc((size_t)N_BAGS * D_FEAT * 4);
  float* partial = (float*)alloc((size_t)N_BAGS * BS_CHUNKS * D_FEAT * 4);
  float* bufA    = (float*)alloc((size_t)N_PATCHES * D_FEAT * 4);  // H1, reused as Sattn
  float* bufB    = (float*)alloc((size_t)N_PATCHES * D_FEAT * 4);  // H2

  // 1. weights f32 -> bf16, transposed to [N,K] for contiguous-K TDM tiles
  cvt_transpose_bf16_kernel<<<(D_IN * D_FEAT + 255) / 256, 256, 0, stream>>>(W1, W1t, D_IN, D_FEAT);
  cvt_transpose_bf16_kernel<<<(D_FEAT * D_FEAT + 255) / 256, 256, 0, stream>>>(W2, W2t, D_FEAT, D_FEAT);
  cvt_transpose_bf16_kernel<<<(D_FEAT * D_ATTN + 255) / 256, 256, 0, stream>>>(Wa1, Wa1t, D_FEAT, D_ATTN);

  const int gm = (N_PATCHES + 127) / 128;
  // 2. H1 = relu(X @ W1 + b1)
  gemm_bias_act_kernel<0, N_PATCHES, D_IN, D_FEAT>
      <<<dim3(gm, D_FEAT / 128), 256, 0, stream>>>(features, W1t, b1, bufA);
  // 3. H2 = relu(H1 @ W2 + b2)
  gemm_bias_act_kernel<0, N_PATCHES, D_FEAT, D_FEAT>
      <<<dim3(gm, D_FEAT / 128), 256, 0, stream>>>(bufA, W2t, b2, bufB);
  // 4. Sattn = tanh(H2 @ Wa1 + ba1)   (reuses bufA)
  gemm_bias_act_kernel<1, N_PATCHES, D_FEAT, D_ATTN>
      <<<dim3(gm, D_ATTN / 128), 256, 0, stream>>>(bufB, Wa1t, ba1, bufA);
  // 5. scores = Sattn @ Wa2 + ba2
  scores_kernel<<<(N_PATCHES + 7) / 8, 256, 0, stream>>>(bufA, Wa2, ba2, scores, N_PATCHES);
  // 6. ragged segment softmax
  offsets_kernel<<<1, 32, 0, stream>>>(bag_sizes, offs, N_BAGS);
  softmax_kernel<<<N_BAGS, 256, 0, stream>>>(scores, offs, attw);
  // 7. bag_emb = segment_sum(w * H2), deterministic two-phase reduction
  bagsum_partial_kernel<<<dim3(N_BAGS, BS_CHUNKS), D_FEAT, 0, stream>>>(bufB, attw, offs, partial);
  bagsum_reduce_kernel<<<N_BAGS, D_FEAT, 0, stream>>>(partial, bag_emb, BS_CHUNKS);
  // 8. out = bag_emb @ Wc + bc
  classifier_kernel<<<1, 64, 0, stream>>>(bag_emb, Wc, bc, out);
}
